// RingDilatedAttentionHilbertOptimizedFixed_47467978555539
// MI455X (gfx1250) — matrix operand order repaired and checked by reading, not compiled, not run on hardware
//
#include <hip/hip_runtime.h>

// ---------------------------------------------------------------------------
// Dilated attention with Hilbert key gather (MI455X / gfx1250, wave32 WMMA).
// Insight: query permutation cancels; only the gathered key subset matters.
// Per group g (5 heads each): S/dil == 2048 keys per segment, flash-attention
// with v_wmma_f32_16x16x32_bf16, f32 accumulation.
// ---------------------------------------------------------------------------

typedef __attribute__((ext_vector_type(16))) __bf16    v16bf;
typedef __attribute__((ext_vector_type(8)))  float     v8f;
typedef __attribute__((ext_vector_type(4)))  float     f32x4;
typedef __attribute__((ext_vector_type(4)))  unsigned  u32x4;

union BFPack {
    unsigned u[8];
    u32x4    q[2];
    v16bf    v;
};

// f32x2 -> packed bf16x2, round-to-nearest-even.
// 2x (v_bfe + v_add3) + one v_perm_b32 (bytes 2,3 of each rounded word).
__device__ __forceinline__ unsigned rne_bf16(float a) {
    unsigned ua = __float_as_uint(a);
    return ua + 0x7FFFu + ((ua >> 16) & 1u);
}
__device__ __forceinline__ unsigned pk_bf16(float a, float b) {
    return __builtin_amdgcn_perm(rne_bf16(b), rne_bf16(a), 0x07060302u);
}
__device__ __forceinline__ unsigned short bf16_1(float a) {
    return (unsigned short)(rne_bf16(a) >> 16);
}

// ---------------------------------------------------------------------------
// Hilbert kv_idx setup: exact transliteration of the reference.
// ---------------------------------------------------------------------------
__device__ __forceinline__ int hilbert_p(int index, int d) {
    int x = 0, y = 0, s = 1;
    while (s < d) {
        int rx = (index >> 1) & 1;
        int ry = (index ^ rx) & 1;
        if (ry == 0) {
            if (rx == 1) { int nx = s - 1 - y, ny = s - 1 - x; x = nx; y = ny; }
            int t = x; x = y; y = t;
        }
        x += s * rx;
        y += s * ry;
        index >>= 2;
        s <<= 1;
    }
    return x * d + y;
}

__global__ void hilbert_kv_kernel(int* __restrict__ kv) {
    __shared__ int cnts[1024];
    const int g   = blockIdx.x;                    // 0..2
    const int S   = (g == 0) ? 2048 : (g == 1) ? 4096 : 8192;
    const int d   = (g == 2) ? 128 : 64;
    const int dil = 1 << g;                        // 1,2,4
    const int Imax = (g == 1) ? 4096 : S;          // min(S, d*d)
    const int t = threadIdx.x;
    const int C = Imax >> 10;                      // per-thread chunk (2/4/8)

    int c = 0;
    for (int j = 0; j < C; ++j)
        c += (hilbert_p(t * C + j, d) < S) ? 1 : 0;
    cnts[t] = c;
    __syncthreads();
    for (int off = 1; off < 1024; off <<= 1) {     // inclusive scan
        int vv = (t >= off) ? cnts[t - off] : 0;
        __syncthreads();
        cnts[t] += vv;
        __syncthreads();
    }
    int r = cnts[t] - c;                           // exclusive rank
    int* kvg = kv + g * 2048;
    for (int j = 0; j < C; ++j) {
        int p = hilbert_p(t * C + j, d);
        if (p < S) {
            if ((r % dil) == 0) kvg[r / dil] = p;
            ++r;
        }
    }
    const int total = cnts[1023];                  // identity padding tail
    for (int j = total + t; j < S; j += 1024)
        if ((j % dil) == 0) kvg[j / dil] = j;
}

// ---------------------------------------------------------------------------
// Head 15 is untouched by the reference groups -> zero it (d_out is poisoned).
// ---------------------------------------------------------------------------
__global__ void zero_head15_kernel(float* __restrict__ out) {
    int i  = blockIdx.x * blockDim.x + threadIdx.x;   // 0 .. 8192*64-1
    int l  = i >> 6, dd = i & 63;
    out[l * 1024 + 15 * 64 + dd] = 0.0f;
}

// ---------------------------------------------------------------------------
// Main flash-attention kernel. 256 threads (8 wave32) per WG; each wave owns
// a 16-query tile; WG cooperatively stages K (row-major bf16) and V^T
// (dim-major bf16) tiles of 32 gathered keys in LDS.
// ---------------------------------------------------------------------------
__global__ __launch_bounds__(256)
void hilbert_dilated_attn_kernel(const float* __restrict__ q,
                                 const float* __restrict__ k,
                                 const float* __restrict__ v,
                                 float* __restrict__ out,
                                 const int* __restrict__ kvws) {
    __shared__ __align__(16) unsigned short Kt[32 * 72];  // 32 keys x 64 dims, stride 72 halves (144B, conflict-padded)
    __shared__ __align__(16) unsigned short Vt[64 * 40];  // 64 dims x 32 keys, stride 40 halves (80B, conflict-padded)

    const int g     = blockIdx.x / 320;
    const int r     = blockIdx.x % 320;
    const int S     = (g == 0) ? 2048 : (g == 1) ? 4096 : 8192;
    const int tiles = S >> 7;                        // 128-query tiles per (seg,head)
    const int sh    = r / tiles;
    const int qt    = r % tiles;
    const int seg   = sh / 5;
    const int h     = g * 5 + (sh - seg * 5);
    const int seg_base = seg * S;
    const int* kv   = kvws + g * 2048;               // 2048 gathered key indices

    const int tid  = threadIdx.x;
    const int lane = tid & 31;
    const int wv   = tid >> 5;
    const int l15  = lane & 15;
    const int hi16 = (lane >> 4) & 1;

    // ---- Q as WMMA B-operand (Q^T), scale (1/8) and log2(e) folded in ----
    const int qrow = seg_base + qt * 128 + wv * 16 + l15;
    const float* qp = q + (size_t)qrow * 1024 + h * 64 + hi16 * 16;
    const float cs = 0.125f * 1.44269504088896340736f;
    BFPack qb0, qb1;
    {
        f32x4 a = *(const f32x4*)(qp + 0),  b = *(const f32x4*)(qp + 4);
        f32x4 c = *(const f32x4*)(qp + 8),  d = *(const f32x4*)(qp + 12);
        qb0.u[0] = pk_bf16(a.x * cs, a.y * cs); qb0.u[1] = pk_bf16(a.z * cs, a.w * cs);
        qb0.u[2] = pk_bf16(b.x * cs, b.y * cs); qb0.u[3] = pk_bf16(b.z * cs, b.w * cs);
        qb0.u[4] = pk_bf16(c.x * cs, c.y * cs); qb0.u[5] = pk_bf16(c.z * cs, c.w * cs);
        qb0.u[6] = pk_bf16(d.x * cs, d.y * cs); qb0.u[7] = pk_bf16(d.z * cs, d.w * cs);
    }
    {
        f32x4 a = *(const f32x4*)(qp + 32), b = *(const f32x4*)(qp + 36);
        f32x4 c = *(const f32x4*)(qp + 40), d = *(const f32x4*)(qp + 44);
        qb1.u[0] = pk_bf16(a.x * cs, a.y * cs); qb1.u[1] = pk_bf16(a.z * cs, a.w * cs);
        qb1.u[2] = pk_bf16(b.x * cs, b.y * cs); qb1.u[3] = pk_bf16(b.z * cs, b.w * cs);
        qb1.u[4] = pk_bf16(c.x * cs, c.y * cs); qb1.u[5] = pk_bf16(c.z * cs, c.w * cs);
        qb1.u[6] = pk_bf16(d.x * cs, d.y * cs); qb1.u[7] = pk_bf16(d.z * cs, d.w * cs);
    }

    // cooperative loader mapping: 256 thr -> 32 keys x 8 dim-chunks
    const int ckey = tid >> 3;
    const int cd   = (tid & 7) << 3;
    const int koff = h * 64 + cd;

    float m = -3.0e38f, ssum = 0.0f;
    const v8f z8 = {0.f, 0.f, 0.f, 0.f, 0.f, 0.f, 0.f, 0.f};
    v8f acc[4] = {z8, z8, z8, z8};                   // O^T tiles: dims 16t..16t+15 x 16 queries

    for (int blk = 0; blk < 64; ++blk) {
        if (blk) __syncthreads();
        // ------- gather + convert K / V^T into LDS -------
        const int kidx = kv[blk * 32 + ckey];
        const size_t rb = (size_t)(seg_base + kidx) * 1024 + koff;
        f32x4 ka = *(const f32x4*)(k + rb);
        f32x4 kb = *(const f32x4*)(k + rb + 4);
        f32x4 va = *(const f32x4*)(v + rb);
        f32x4 vb = *(const f32x4*)(v + rb + 4);
        if (blk + 1 < 64) {                          // global_prefetch_b8 next gathered rows
            int nidx = kv[(blk + 1) * 32 + ckey];
            size_t nb = (size_t)(seg_base + nidx) * 1024 + koff;
            __builtin_prefetch((const void*)(k + nb), 0, 0);
            __builtin_prefetch((const void*)(v + nb), 0, 0);
        }
        u32x4 kp;
        kp.x = pk_bf16(ka.x, ka.y); kp.y = pk_bf16(ka.z, ka.w);
        kp.z = pk_bf16(kb.x, kb.y); kp.w = pk_bf16(kb.z, kb.w);
        *(u32x4*)&Kt[ckey * 72 + cd] = kp;           // 16B aligned (144B row stride)
        Vt[(cd + 0) * 40 + ckey] = bf16_1(va.x);
        Vt[(cd + 1) * 40 + ckey] = bf16_1(va.y);
        Vt[(cd + 2) * 40 + ckey] = bf16_1(va.z);
        Vt[(cd + 3) * 40 + ckey] = bf16_1(va.w);
        Vt[(cd + 4) * 40 + ckey] = bf16_1(vb.x);
        Vt[(cd + 5) * 40 + ckey] = bf16_1(vb.y);
        Vt[(cd + 6) * 40 + ckey] = bf16_1(vb.z);
        Vt[(cd + 7) * 40 + ckey] = bf16_1(vb.w);
        __syncthreads();

        // ------- S^T = K_tile . Q^T  (two 16-key subtiles, K-dim=64 via 2 steps) -------
        v8f t0 = z8, t1 = z8;
        #pragma unroll
        for (int kk = 0; kk < 2; ++kk) {
            const unsigned short* b0 = &Kt[l15 * 72 + kk * 32 + hi16 * 16];
            BFPack a0, a1;
            a0.q[0] = *(const u32x4*)b0;
            a0.q[1] = *(const u32x4*)(b0 + 8);            // +16B: dims +8..15 (contiguous k order)
            a1.q[0] = *(const u32x4*)(b0 + 16 * 72);      // keys 16..31 subtile
            a1.q[1] = *(const u32x4*)(b0 + 16 * 72 + 8);
            const v16bf qv = (kk == 0) ? qb0.v : qb1.v;
            t0 = __builtin_amdgcn_wmma_f32_16x16x32_bf16(false, a0.v, false, qv, (short)0, t0, false, false);
            t1 = __builtin_amdgcn_wmma_f32_16x16x32_bf16(false, a1.v, false, qv, (short)0, t1, false, false);
        }

        // ------- online softmax (exp2 domain; lane q & q+16 split the 32 keys) -------
        float lmax = t0[0];
        #pragma unroll
        for (int i = 1; i < 8; ++i) lmax = fmaxf(lmax, t0[i]);
        #pragma unroll
        for (int i = 0; i < 8; ++i) lmax = fmaxf(lmax, t1[i]);
        float bmax = fmaxf(lmax, __shfl_xor(lmax, 16, 32));
        float mnew = fmaxf(m, bmax);
        float scl  = __builtin_amdgcn_exp2f(m - mnew);
        m = mnew;
        ssum *= scl;
        #pragma unroll
        for (int t = 0; t < 4; ++t)
            #pragma unroll
            for (int i = 0; i < 8; ++i) acc[t][i] *= scl;

        float e0[8], e1[8], ls = 0.f;
        #pragma unroll
        for (int i = 0; i < 8; ++i) { e0[i] = __builtin_amdgcn_exp2f(t0[i] - mnew); ls += e0[i]; }
        #pragma unroll
        for (int i = 0; i < 8; ++i) { e1[i] = __builtin_amdgcn_exp2f(t1[i] - mnew); ls += e1[i]; }
        ssum += ls;

        // P^T as B-operand, built purely per-lane (key-index permutation trick:
        // logical K {0..7}->keys0..7, {8..15}->16..23, {16..23}->8..15, {24..31}->24..31;
        // the V^T A-operand below uses the same mapping, so products align)
        BFPack pb;
        #pragma unroll
        for (int j = 0; j < 4; ++j) pb.u[j]     = pk_bf16(e0[2 * j], e0[2 * j + 1]);
        #pragma unroll
        for (int j = 0; j < 4; ++j) pb.u[4 + j] = pk_bf16(e1[2 * j], e1[2 * j + 1]);

        // ------- O^T += V^T . P^T (4 dim-tiles, K-dim = 32 keys) -------
        #pragma unroll
        for (int t = 0; t < 4; ++t) {
            const unsigned short* vb0 = &Vt[(16 * t + l15) * 40 + hi16 * 8];
            BFPack va_;
            va_.q[0] = *(const u32x4*)vb0;        // keys 8h..8h+7, then 16+8h..16+8h+7
            va_.q[1] = *(const u32x4*)(vb0 + 16);
            acc[t] = __builtin_amdgcn_wmma_f32_16x16x32_bf16(false, va_.v, false, pb.v, (short)0, acc[t], false, false);
        }
    }

    // ------- epilogue: combine lane-halves' sums, normalize, store -------
    float osum = ssum + __shfl_xor(ssum, 16, 32);
    float rcp  = 1.0f / osum;
    float* op = out + (size_t)qrow * 1024 + h * 64 + hi16 * 8;
    #pragma unroll
    for (int t = 0; t < 4; ++t) {
        f32x4 s0, s1;
        s0.x = acc[t][0] * rcp; s0.y = acc[t][1] * rcp;
        s0.z = acc[t][2] * rcp; s0.w = acc[t][3] * rcp;
        s1.x = acc[t][4] * rcp; s1.y = acc[t][5] * rcp;
        s1.z = acc[t][6] * rcp; s1.w = acc[t][7] * rcp;
        *(f32x4*)(op + t * 16)     = s0;
        *(f32x4*)(op + t * 16 + 4) = s1;
    }
}

// ---------------------------------------------------------------------------
extern "C" void kernel_launch(void* const* d_in, const int* in_sizes, int n_in,
                              void* d_out, int out_size, void* d_ws, size_t ws_size,
                              hipStream_t stream) {
    const float* q = (const float*)d_in[0];
    const float* k = (const float*)d_in[1];
    const float* v = (const float*)d_in[2];
    float* out = (float*)d_out;
    int* kv = (int*)d_ws;                           // 3 * 2048 ints

    hilbert_kv_kernel<<<3, 1024, 0, stream>>>(kv);
    zero_head15_kernel<<<2048, 256, 0, stream>>>(out);
    hilbert_dilated_attn_kernel<<<960, 256, 0, stream>>>(q, k, v, out, kv);

    (void)in_sizes; (void)n_in; (void)out_size; (void)ws_size;
}
